// GNN_3977139716839
// MI455X (gfx1250) — matrix operand order repaired and checked by reading, not compiled, not run on hardware
//
#include <hip/hip_runtime.h>
#include <hip/hip_bf16.h>
#include <math.h>

// MI455X / gfx1250, wave32. GEMMs via v_wmma_f32_16x16x32_bf16.
// Weights are pre-packed once per call into bf16 WMMA B-operand tile layout,
// so the GEMM hot loop is: contiguous async 8KB B-slab copy to LDS (ASYNCcnt)
// + A fragment loads + 2 WMMAs per K-step, double-buffered, one barrier/step.
// Assumes: N divisible by 32 (20000 ok), K in {128,256,512}, O multiple of 128.
// Scratch use: ~147 MB in d_ws.

typedef __attribute__((ext_vector_type(16))) __bf16 v16bf;
typedef __attribute__((ext_vector_type(8)))  float  v8f;
typedef __attribute__((ext_vector_type(4)))  int    v4i;

#define HEADS 8
#define DHEAD 32
#define GROUPS 64
#define HIDDEN 128

#if defined(__HIP_DEVICE_COMPILE__) && \
    __has_builtin(__builtin_amdgcn_global_load_async_to_lds_b128) && \
    __has_builtin(__builtin_amdgcn_s_wait_asynccnt)
#define USE_ASYNC_LDS 1
typedef __attribute__((address_space(1))) v4i* gv4i_p;  // global int4*
typedef __attribute__((address_space(3))) v4i* lv4i_p;  // LDS int4*
#else
#define USE_ASYNC_LDS 0
#endif

// ---------- monotone float<->uint mapping for atomicMax on floats ----------
static __device__ __forceinline__ unsigned fkey(float x) {
  unsigned u = __float_as_uint(x);
  return (u & 0x80000000u) ? ~u : (u | 0x80000000u);
}
static __device__ __forceinline__ float funkey(unsigned u) {
  u = (u & 0x80000000u) ? (u & 0x7fffffffu) : ~u;
  return __uint_as_float(u);
}

// ---------- utility kernels ----------
__global__ void k_zero_f(float* p, size_t n) {
  size_t i = (size_t)blockIdx.x * blockDim.x + threadIdx.x;
  size_t s = (size_t)gridDim.x * blockDim.x;
  for (; i < n; i += s) p[i] = 0.f;
}
__global__ void k_zero_u(unsigned* p, size_t n) {
  size_t i = (size_t)blockIdx.x * blockDim.x + threadIdx.x;
  size_t s = (size_t)gridDim.x * blockDim.x;
  for (; i < n; i += s) p[i] = 0u;
}

__global__ void k_deg(const int* __restrict__ dst, float* __restrict__ cnt, int E) {
  int e = blockIdx.x * blockDim.x + threadIdx.x;
  if (e < E) atomicAdd(&cnt[dst[e]], 1.0f);
}

// scatter-sum of per-node features along edges: sum[dst] += x[src]
__global__ void k_scatter_feat(const float* __restrict__ x, const int* __restrict__ src,
                               const int* __restrict__ dst, float* __restrict__ sum,
                               int F, int total) {
  int i = blockIdx.x * blockDim.x + threadIdx.x;
  if (i >= total) return;
  int e = i / F;
  int c = i % F;
  atomicAdd(&sum[dst[e] * F + c], x[src[e] * F + c]);
}

__global__ void k_mean(float* __restrict__ sum, const float* __restrict__ cnt,
                       int F, int total) {
  int i = blockIdx.x * blockDim.x + threadIdx.x;
  if (i >= total) return;
  int n = i / F;
  sum[i] = sum[i] * __builtin_amdgcn_rcpf(fmaxf(cnt[n], 1.0f));
}

// ---------- weight pre-pack: fp32 W[K,O] -> bf16 WMMA B-operand tiles ----------
// Packed index: (((kk/32)*(O/16) + ntile)*32 + l)*16 + s
//   holds W[(kk/32)*32 + (l&16) + s, ntile*16 + (l&15)].
// For a GEMM block at (kk, nbase): its 8 n-tiles are CONTIGUOUS 8KB in P.
__global__ void k_pack_w(const float* __restrict__ W, __bf16* __restrict__ P,
                         int K, int O) {
  int idx = blockIdx.x * blockDim.x + threadIdx.x;
  if (idx >= K * O) return;
  int s    = idx & 15;
  int l    = (idx >> 4) & 31;
  int rest = idx >> 9;             // c*(O/16) + ntile
  int nt   = rest % (O >> 4);
  int c    = rest / (O >> 4);
  int gk   = c * 32 + (l & 16) + s;
  int gn   = nt * 16 + (l & 15);
  P[idx] = (__bf16)W[gk * O + gn];
}

// ---------- WMMA GEMM: C[M,O] = act(A[M,K] @ W[K,O] + bias + add) ----------
// Block: 256 threads = 8 waves. Block tile 32(M) x 128(N).
// Wave w: m-strip mt = w&1 (16 rows), n-group ng = w>>1 -> n-tiles 2*ng, 2*ng+1.
// Each wave keeps 2 f32 accumulators; one A fragment feeds 2 WMMAs.
// B comes from pre-packed bf16 tiles -> staging is a contiguous 8KB copy,
// issued as GLOBAL_LOAD_ASYNC_TO_LDS_B128 (ASYNCcnt) when available.
// grid.x = M/32, grid.y = O/128.
template <int ACT, bool HAS_ADD>
__global__ __launch_bounds__(256)
void k_gemm_wmma(const float* __restrict__ A, const __bf16* __restrict__ Wpk,
                 const float* __restrict__ bias, const float* __restrict__ add,
                 float* __restrict__ C, int K, int O) {
  __shared__ __align__(32) __bf16 b_lds[2][8][32][16];  // 16 KB

  const int tid  = threadIdx.x;
  const int lane = tid & 31;
  const int wv   = tid >> 5;
  const int mt   = wv & 1;        // m-strip in block
  const int ng   = wv >> 1;       // n-group (2 n-tiles)
  const int nt0  = ng * 2;
  const int nt1  = nt0 + 1;
  const int m0   = blockIdx.x * 32 + mt * 16;
  const int nbase = blockIdx.y * 128;
  const int row  = m0 + (lane & 15);
  const int hi   = lane >> 4;     // 0: lanes 0-15, 1: lanes 16-31
  const int ntPerO = O >> 4;

  v8f acc0 = {};
  v8f acc1 = {};

  // Copy one contiguous 8KB packed B slab (32K x 128N) into LDS buffer `buf`.
  auto stage = [&](int buf, int kk) {
    int base = (((kk >> 5) * ntPerO) + (nbase >> 4)) << 9;  // bf16 elements
#if USE_ASYNC_LDS
    const __bf16* g = Wpk + base;
    __bf16* l = &b_lds[buf][0][0][0];
    __builtin_amdgcn_global_load_async_to_lds_b128(
        (gv4i_p)(g + tid * 8), (lv4i_p)(l + tid * 8), 0, 0);
    __builtin_amdgcn_global_load_async_to_lds_b128(
        (gv4i_p)(g + 2048 + tid * 8), (lv4i_p)(l + 2048 + tid * 8), 0, 0);
#else
    const uint4* s4 = (const uint4*)(Wpk + base);
    uint4* d4 = (uint4*)&b_lds[buf][0][0][0];
    d4[tid]       = s4[tid];
    d4[tid + 256] = s4[tid + 256];
#endif
  };
  auto stage_wait = [&]() {
#if USE_ASYNC_LDS
    __builtin_amdgcn_s_wait_asynccnt(0);
#endif
  };

  // One K-step (32) of compute from LDS buffer `buf`.
  auto compute = [&](int buf, int kk) {
    const float* Ar = A + (row * K + kk + hi * 16);  // 32-bit index math
    __builtin_prefetch(Ar + 64, 0, 3);               // global_prefetch of next A chunk
    float4 a0 = *(const float4*)(Ar + 0);
    float4 a1 = *(const float4*)(Ar + 4);
    float4 a2 = *(const float4*)(Ar + 8);
    float4 a3 = *(const float4*)(Ar + 12);
    v16bf av;
    av[0]  = (__bf16)a0.x; av[1]  = (__bf16)a0.y; av[2]  = (__bf16)a0.z; av[3]  = (__bf16)a0.w;
    av[4]  = (__bf16)a1.x; av[5]  = (__bf16)a1.y; av[6]  = (__bf16)a1.z; av[7]  = (__bf16)a1.w;
    av[8]  = (__bf16)a2.x; av[9]  = (__bf16)a2.y; av[10] = (__bf16)a2.z; av[11] = (__bf16)a2.w;
    av[12] = (__bf16)a3.x; av[13] = (__bf16)a3.y; av[14] = (__bf16)a3.z; av[15] = (__bf16)a3.w;

    v16bf bv0 = *(const v16bf*)(&b_lds[buf][nt0][lane][0]);
    v16bf bv1 = *(const v16bf*)(&b_lds[buf][nt1][lane][0]);

    acc0 = __builtin_amdgcn_wmma_f32_16x16x32_bf16(false, av, false, bv0,
                                                   (short)0, acc0, false, false);
    acc1 = __builtin_amdgcn_wmma_f32_16x16x32_bf16(false, av, false, bv1,
                                                   (short)0, acc1, false, false);
  };

  stage(0, 0);
  stage_wait();
  __syncthreads();
  for (int kk = 0; kk < K; kk += 64) {      // K is a multiple of 64 here
    if (kk + 32 < K) stage(1, kk + 32);
    compute(0, kk);
    stage_wait();
    __syncthreads();
    if (kk + 64 < K) stage(0, kk + 64);
    if (kk + 32 < K) compute(1, kk + 32);
    stage_wait();
    __syncthreads();
  }

  // Epilogue. C/D layout: VGPR r -> M = r + (lane<16 ? 0 : 8), N = lane&15.
  const int col0 = nbase + nt0 * 16 + (lane & 15);
  const int col1 = nbase + nt1 * 16 + (lane & 15);
  const int rb   = m0 + hi * 8;
  const float bb0 = bias ? bias[col0] : 0.0f;
  const float bb1 = bias ? bias[col1] : 0.0f;
#pragma unroll
  for (int r = 0; r < 8; ++r) {
    int rowoff = (rb + r) * O;               // fits in 32 bits (max ~10.2M)
    float v0 = acc0[r] + bb0;
    float v1 = acc1[r] + bb1;
    if (HAS_ADD) { v0 += add[rowoff + col0]; v1 += add[rowoff + col1]; }
    if (ACT == 1) { v0 = fmaxf(v0, 0.0f); v1 = fmaxf(v1, 0.0f); }
    else if (ACT == 2) {
      v0 = __builtin_amdgcn_rcpf(1.0f + __expf(-v0));
      v1 = __builtin_amdgcn_rcpf(1.0f + __expf(-v1));
    }
    C[rowoff + col0] = v0;
    C[rowoff + col1] = v1;
  }
}

// ---------- TransformerConv edge kernels ----------
__global__ void k_logits(const float* __restrict__ q, const float* __restrict__ kmat,
                         const int* __restrict__ src, const int* __restrict__ dst,
                         float* __restrict__ logits, int E) {
  int i = blockIdx.x * blockDim.x + threadIdx.x;
  if (i >= E * HEADS) return;
  int e = i >> 3;
  int h = i & 7;
  const float4* qp = (const float4*)(q    + dst[e] * 256 + h * DHEAD);
  const float4* kp = (const float4*)(kmat + src[e] * 256 + h * DHEAD);
  float s = 0.f;
#pragma unroll
  for (int j = 0; j < 8; ++j) {
    float4 a = qp[j], b = kp[j];
    s += a.x * b.x + a.y * b.y + a.z * b.z + a.w * b.w;
  }
  logits[i] = s * 0.17677669529663687f;  // 1/sqrt(32)
}

__global__ void k_segmax(const float* __restrict__ logits, const int* __restrict__ dst,
                         unsigned* __restrict__ mbits, int E) {
  int i = blockIdx.x * blockDim.x + threadIdx.x;
  if (i >= E * HEADS) return;
  int e = i >> 3, h = i & 7;
  atomicMax(&mbits[dst[e] * HEADS + h], fkey(logits[i]));
}

__global__ void k_expsum(const float* __restrict__ logits, const unsigned* __restrict__ mbits,
                         const int* __restrict__ dst, float* __restrict__ pexp,
                         float* __restrict__ denom, int E) {
  int i = blockIdx.x * blockDim.x + threadIdx.x;
  if (i >= E * HEADS) return;
  int e = i >> 3, h = i & 7;
  float m = funkey(mbits[dst[e] * HEADS + h]);
  float p = __expf(logits[i] - m);
  pexp[i] = p;
  atomicAdd(&denom[dst[e] * HEADS + h], p);
}

__global__ void k_scatter_v(const float* __restrict__ pexp, const float* __restrict__ denom,
                            const float* __restrict__ v, const int* __restrict__ src,
                            const int* __restrict__ dst, float* __restrict__ agg, int E) {
  int i = blockIdx.x * blockDim.x + threadIdx.x;
  if (i >= E * HEADS) return;
  int e = i >> 3, h = i & 7;
  int d = dst[e], s = src[e];
  float scale = pexp[i] * __builtin_amdgcn_rcpf(denom[d * HEADS + h]);
  const float* vp = v + s * 256 + h * DHEAD;
  float* ap = agg + d * 256 + h * DHEAD;
#pragma unroll
  for (int j = 0; j < DHEAD; ++j) atomicAdd(&ap[j], scale * vp[j]);
}

// ---------- global max pool ----------
__global__ void k_pool(const float* __restrict__ h, const int* __restrict__ batch,
                       unsigned* __restrict__ poolbits, int total) {
  int i = blockIdx.x * blockDim.x + threadIdx.x;
  if (i >= total) return;
  int n = i / HIDDEN;
  int c = i % HIDDEN;
  atomicMax(&poolbits[batch[n] * HIDDEN + c], fkey(h[i]));
}

__global__ void k_pool_out(const unsigned* __restrict__ poolbits, float* __restrict__ out, int n) {
  int i = blockIdx.x * blockDim.x + threadIdx.x;
  if (i < n) out[i] = funkey(poolbits[i]);
}

// ---------- launch ----------
extern "C" void kernel_launch(void* const* d_in, const int* in_sizes, int n_in,
                              void* d_out, int out_size, void* d_ws, size_t ws_size,
                              hipStream_t stream) {
  const float* x     = (const float*)d_in[0];
  const int*   ei    = (const int*)d_in[1];
  const int*   batch = (const int*)d_in[2];
  const float* s1_wl = (const float*)d_in[3];
  const float* s1_bl = (const float*)d_in[4];
  const float* s1_wr = (const float*)d_in[5];
  const float* s2_wl = (const float*)d_in[6];
  const float* s2_bl = (const float*)d_in[7];
  const float* s2_wr = (const float*)d_in[8];
  const float* t1_wq = (const float*)d_in[9];  const float* t1_bq = (const float*)d_in[10];
  const float* t1_wk = (const float*)d_in[11]; const float* t1_bk = (const float*)d_in[12];
  const float* t1_wv = (const float*)d_in[13]; const float* t1_bv = (const float*)d_in[14];
  const float* t1_ws = (const float*)d_in[15]; const float* t1_bs = (const float*)d_in[16];
  const float* t2_wq = (const float*)d_in[17]; const float* t2_bq = (const float*)d_in[18];
  const float* t2_wk = (const float*)d_in[19]; const float* t2_bk = (const float*)d_in[20];
  const float* t2_wv = (const float*)d_in[21]; const float* t2_bv = (const float*)d_in[22];
  const float* t2_ws = (const float*)d_in[23]; const float* t2_bs = (const float*)d_in[24];
  const float* p1_w  = (const float*)d_in[25]; const float* p1_b  = (const float*)d_in[26];
  const float* p2_w  = (const float*)d_in[27]; const float* p2_b  = (const float*)d_in[28];

  const int N = in_sizes[2];       // 20000
  const int E = in_sizes[1] / 2;   // 320000
  const int* src = ei;
  const int* dst = ei + E;

  // workspace layout (fp32 elements; every section is 16B aligned)
  float* w = (float*)d_ws;
  float*    cnt      = w;             w += N;
  unsigned* mbits    = (unsigned*)w;  w += (size_t)N * HEADS;
  float*    denom    = w;             w += (size_t)N * HEADS;
  float*    logits   = w;             w += (size_t)E * HEADS;
  float*    pexp     = w;             w += (size_t)E * HEADS;
  unsigned* poolbits = (unsigned*)w;  w += (size_t)GROUPS * HIDDEN;
  float* S[6];
  for (int i = 0; i < 6; ++i) { S[i] = w; w += (size_t)N * 256; }
  // Note: S[0] and S[1] are contiguous -> N x 512 region for proj layer 1.
  __bf16* pkcur = (__bf16*)w;   // packed-weight region (~1.6 MB)

  auto zf = [&](float* p, size_t n) { k_zero_f<<<1024, 256, 0, stream>>>(p, n); };
  auto zu = [&](unsigned* p, size_t n) { k_zero_u<<<1024, 256, 0, stream>>>(p, n); };

  // pre-pack a weight into bf16 B-operand tiles; returns packed pointer
  auto pack = [&](const float* W, int K, int O) -> const __bf16* {
    __bf16* p = pkcur; pkcur += (size_t)K * O;
    k_pack_w<<<(K * O + 255) / 256, 256, 0, stream>>>(W, p, K, O);
    return p;
  };
  const __bf16* P_s1wl = pack(s1_wl, 128, 128);
  const __bf16* P_s1wr = pack(s1_wr, 128, 128);
  const __bf16* P_s2wl = pack(s2_wl, 128, 256);
  const __bf16* P_s2wr = pack(s2_wr, 128, 256);
  const __bf16* P_t1wq = pack(t1_wq, 256, 256);
  const __bf16* P_t1wk = pack(t1_wk, 256, 256);
  const __bf16* P_t1wv = pack(t1_wv, 256, 256);
  const __bf16* P_t1ws = pack(t1_ws, 256, 256);
  const __bf16* P_t2wq = pack(t2_wq, 256, 256);
  const __bf16* P_t2wk = pack(t2_wk, 256, 256);
  const __bf16* P_t2wv = pack(t2_wv, 256, 256);
  const __bf16* P_t2ws = pack(t2_ws, 256, 256);
  const __bf16* P_p1w  = pack(p1_w, 256, 512);
  const __bf16* P_p2w  = pack(p2_w, 512, 128);

  auto gemm = [&](const float* A, const __bf16* Wpk, const float* b, const float* add,
                  float* C, int K, int O, int act) {
    dim3 g(N / 32, O / 128);
    if (add) {
      if (act == 1) k_gemm_wmma<1, true><<<g, 256, 0, stream>>>(A, Wpk, b, add, C, K, O);
      else          k_gemm_wmma<0, true><<<g, 256, 0, stream>>>(A, Wpk, b, add, C, K, O);
    } else {
      if (act == 0)      k_gemm_wmma<0, false><<<g, 256, 0, stream>>>(A, Wpk, b, add, C, K, O);
      else if (act == 1) k_gemm_wmma<1, false><<<g, 256, 0, stream>>>(A, Wpk, b, add, C, K, O);
      else               k_gemm_wmma<2, false><<<g, 256, 0, stream>>>(A, Wpk, b, add, C, K, O);
    }
  };

  const int egrid = (E * HEADS + 255) / 256;

  auto sage = [&](const float* hin, int F, const __bf16* wl, const float* bl,
                  const __bf16* wr, int O, float* hout) {
    zf(S[0], (size_t)N * F);
    int te = E * F;
    k_scatter_feat<<<(te + 255) / 256, 256, 0, stream>>>(hin, src, dst, S[0], F, te);
    int tn = N * F;
    k_mean<<<(tn + 255) / 256, 256, 0, stream>>>(S[0], cnt, F, tn);
    gemm(hin, wr, nullptr, nullptr, S[1], F, O, 0);          // x @ wr
    gemm(S[0], wl, bl, S[1], hout, F, O, 1);                 // relu(mean@wl + bl + xwr)
  };

  auto tconv = [&](const float* hin,
                   const __bf16* wq, const float* bq, const __bf16* wk_, const float* bk_,
                   const __bf16* wv_, const float* bv_, const __bf16* ws_, const float* bs_,
                   float* hout) {
    gemm(hin, wq,  bq,  nullptr, S[0], 256, 256, 0);  // q
    gemm(hin, wk_, bk_, nullptr, S[1], 256, 256, 0);  // k
    gemm(hin, wv_, bv_, nullptr, S[2], 256, 256, 0);  // v
    zu(mbits, (size_t)N * HEADS);
    zf(denom, (size_t)N * HEADS);
    zf(S[4], (size_t)N * 256);
    k_logits   <<<egrid, 256, 0, stream>>>(S[0], S[1], src, dst, logits, E);
    k_segmax   <<<egrid, 256, 0, stream>>>(logits, dst, mbits, E);
    k_expsum   <<<egrid, 256, 0, stream>>>(logits, mbits, dst, pexp, denom, E);
    k_scatter_v<<<egrid, 256, 0, stream>>>(pexp, denom, S[2], src, dst, S[4], E);
    gemm(hin, ws_, bs_, S[4], hout, 256, 256, 1);     // relu(agg + x@ws + bs)
  };

  // degree (shared by both SAGE layers)
  zf(cnt, (size_t)N);
  k_deg<<<(E + 255) / 256, 256, 0, stream>>>(dst, cnt, E);

  sage(x,    128, P_s1wl, s1_bl, P_s1wr, 128, S[2]);  // h1 -> S2
  sage(S[2], 128, P_s2wl, s2_bl, P_s2wr, 256, S[3]);  // h2 -> S3

  tconv(S[3], P_t1wq, t1_bq, P_t1wk, t1_bk, P_t1wv, t1_bv, P_t1ws, t1_bs, S[5]);  // h3 -> S5
  tconv(S[5], P_t2wq, t2_bq, P_t2wk, t2_bk, P_t2wv, t2_bv, P_t2ws, t2_bs, S[3]);  // h4 -> S3

  // proj MLP: relu(h@p1_w + p1_b) -> sigmoid(@p2_w + p2_b)
  gemm(S[3], P_p1w, p1_b, nullptr, S[0], 256, 512, 1);   // into S0..S1 (N x 512)
  gemm(S[0], P_p2w, p2_b, nullptr, S[2], 512, 128, 2);   // N x 128

  // global max pool
  zu(poolbits, (size_t)GROUPS * HIDDEN);
  int tp = N * HIDDEN;
  k_pool<<<(tp + 255) / 256, 256, 0, stream>>>(S[2], batch, poolbits, tp);
  k_pool_out<<<(GROUPS * HIDDEN + 255) / 256, 256, 0, stream>>>(poolbits, (float*)d_out,
                                                                GROUPS * HIDDEN);
}
